// ED_549755814486
// MI455X (gfx1250) — matrix operand order repaired and checked by reading, not compiled, not run on hardware
//
#include <hip/hip_runtime.h>

typedef _Float16 f16;
typedef __attribute__((ext_vector_type(16))) _Float16 v16h;
typedef __attribute__((ext_vector_type(8)))  _Float16 v8h;
typedef __attribute__((ext_vector_type(4)))  _Float16 v4h;
typedef __attribute__((ext_vector_type(8)))  float    v8f;

#define B_    32
#define C_    128
#define HW_   4096
#define CHW_  (C_*HW_)      // 524288
#define PTILE 128
#define XS    136           // padded LDS row stride in halves (conflict-free)

// workspace layout (float offsets; f16 regions counted as half-sized float counts)
#define WS_CENT 0
#define WS_SIM  (WS_CENT + B_*C_)      // 4096
#define WS_ATT  (WS_SIM  + B_*HW_)     // 4096 + 131072
#define WS_WH   (WS_ATT  + B_*HW_)     // + 131072 -> 266240 floats ; f16 weights 256x128 = 16384 floats
#define WS_VT   (WS_WH   + 256*C_/2)   // 282624 floats -> byte 1130496 (16B aligned); f16 v, 32 MB

// ---------------------------------------------------------------------------
// Kernel 0: convert k/v weight rows (o = 128..383, contiguous) to f16 once.
// ---------------------------------------------------------------------------
__global__ void k_wconv(const float* __restrict__ W, f16* __restrict__ Wh) {
    const int i = (blockIdx.x*256 + threadIdx.x)*4;   // 32768 elements total
    const float4 w = *(const float4*)(W + (size_t)C_*C_ + i);
    v4h h; h[0]=(f16)w.x; h[1]=(f16)w.y; h[2]=(f16)w.z; h[3]=(f16)w.w;
    *(v4h*)(Wh + i) = h;
}

// ---------------------------------------------------------------------------
// Kernel 1: cent[b][c] = sum_c' Wqkv[c][c'] * fmap[b][c'][2047]  (q rows, f32 exact)
// ---------------------------------------------------------------------------
__global__ void k_cent(const float* __restrict__ fmap, const float* __restrict__ W,
                       float* __restrict__ cent) {
    __shared__ float col[C_];
    const int b = blockIdx.x, t = threadIdx.x;
    col[t] = fmap[(size_t)b*CHW_ + (size_t)t*HW_ + (HW_-1)/2];
    __syncthreads();
    float s = 0.f;
    const float* wr = W + (size_t)t*C_;
    #pragma unroll 8
    for (int c = 0; c < C_; ++c) s += wr[c]*col[c];
    cent[b*C_ + t] = s;
}

// ---------------------------------------------------------------------------
// Kernel 2: WMMA GEMM for k/v rows over a 128-position tile, fused ||cent-k||
// epilogue (ds_add_f32) and transposed f16 v store.
//   8 waves: wave w owns o-tiles {2w,2w+1} (32 channels) x all 8 p-tiles.
// ---------------------------------------------------------------------------
__global__ __launch_bounds__(256) void k_qkv(const float* __restrict__ fmap,
                                             const f16*   __restrict__ Wh,
                                             const float* __restrict__ cent,
                                             float* __restrict__ sim,
                                             f16*   __restrict__ vt) {
    __shared__ f16   Xh[PTILE*XS];   // fmap tile, Xh[p][c], f16, ~34 KB
    __shared__ float centS[C_];
    __shared__ float distS[PTILE];

    const int b    = blockIdx.y;
    const int p0   = blockIdx.x * PTILE;
    const int tid  = threadIdx.x;
    const int wave = tid >> 5;
    const int lane = tid & 31;
    const int m    = lane & 15;     // A: row M  /  B: col N
    const int hi   = lane >> 4;     // lane-half selects K sub-range

    // ---- stage X tile (f32 -> f16), transposed into Xh[p][c] ----
    {
        const int p  = tid & 127;
        const int c0 = tid >> 7;
        const float* src = fmap + (size_t)b*CHW_ + p0 + p;
        #pragma unroll 4
        for (int c = c0; c < C_; c += 2)
            Xh[p*XS + c] = (f16)src[(size_t)c*HW_];
    }
    if (tid < C_)    centS[tid] = cent[b*C_ + tid];
    if (tid < PTILE) distS[tid] = 0.f;
    __syncthreads();

    v8f acc[2][8];
    #pragma unroll
    for (int j = 0; j < 2; ++j)
        #pragma unroll
        for (int pt = 0; pt < 8; ++pt) {
            v8f z = {0.f,0.f,0.f,0.f,0.f,0.f,0.f,0.f};
            acc[j][pt] = z;
        }

    // ---- main K loop: 4 chunks of K=32, 64 WMMAs per wave ----
    #pragma unroll
    for (int ko = 0; ko < 4; ++ko) {
        // A fragments straight from pre-converted f16 weights (L2-resident).
        // A k order used in this code: halves h -> K = 16*hi + h (matches B)
        v16h aF[2];
        #pragma unroll
        for (int j = 0; j < 2; ++j) {
            const v8h* wr = (const v8h*)(Wh + (size_t)(32*wave + 16*j + m)*C_
                                            + ko*32 + hi*16);
            v8h a0 = wr[0];          // K = 16*hi + 0..7
            v8h a1 = wr[1];          // K = 16*hi + 8..15
            aF[j] = __builtin_shufflevector(a0, a1, 0,1,2,3,4,5,6,7,8,9,10,11,12,13,14,15);
        }
        // B fragments from LDS: lane n = m, halves h -> K = 16*hi + h (16B aligned)
        #pragma unroll
        for (int pt = 0; pt < 8; ++pt) {
            const v8h* bs = (const v8h*)&Xh[(pt*16 + m)*XS + ko*32 + hi*16];
            v8h b0 = bs[0], b1 = bs[1];
            v16h bF = __builtin_shufflevector(b0, b1, 0,1,2,3,4,5,6,7,8,9,10,11,12,13,14,15);
            acc[0][pt] = __builtin_amdgcn_wmma_f32_16x16x32_f16(
                false, aF[0], false, bF, (short)0, acc[0][pt], false, false);
            acc[1][pt] = __builtin_amdgcn_wmma_f32_16x16x32_f16(
                false, aF[1], false, bF, (short)0, acc[1][pt], false, false);
        }
    }

    // ---- epilogue ----
    // D layout: VGPR i, lane-half hi -> channel offset (i + 8*hi), lane n = column p.
    if (wave < 4) {
        // k channels [32w, 32w+32): accumulate squared distance to cent
        #pragma unroll
        for (int j = 0; j < 2; ++j)
            #pragma unroll
            for (int pt = 0; pt < 8; ++pt) {
                float part = 0.f;
                #pragma unroll
                for (int i = 0; i < 8; ++i) {
                    float d = centS[32*wave + 16*j + 8*hi + i] - acc[j][pt][i];
                    part += d*d;
                }
                atomicAdd(&distS[pt*16 + m], part);   // ds_add_f32
            }
    } else {
        // v channels: store transposed f16 v_t[b][p][c] (8 contiguous halves = b128)
        const int cw = 32*(wave - 4);
        #pragma unroll
        for (int j = 0; j < 2; ++j)
            #pragma unroll
            for (int pt = 0; pt < 8; ++pt) {
                f16* dst = vt + ((size_t)b*HW_ + p0 + pt*16 + m)*C_ + cw + 16*j + 8*hi;
                v8h hv;
                #pragma unroll
                for (int i = 0; i < 8; ++i) hv[i] = (f16)acc[j][pt][i];
                *(v8h*)dst = hv;
            }
    }
    __syncthreads();
    if (tid < PTILE) {
        float d = sqrtf(distS[tid]);
        sim[(size_t)b*HW_ + p0 + tid] = 1.f/(1.f + d);
    }
}

// ---------------------------------------------------------------------------
// Kernel 3: softmax over 4096 positions per batch
// ---------------------------------------------------------------------------
__global__ __launch_bounds__(256) void k_softmax(const float* __restrict__ sim,
                                                 float* __restrict__ att) {
    __shared__ float red[256];
    const int b = blockIdx.x, t = threadIdx.x;
    float v[16];
    float mx = -1e30f;
    #pragma unroll
    for (int r = 0; r < 16; ++r) { v[r] = sim[(size_t)b*HW_ + r*256 + t]; mx = fmaxf(mx, v[r]); }
    red[t] = mx; __syncthreads();
    for (int s = 128; s > 0; s >>= 1) { if (t < s) red[t] = fmaxf(red[t], red[t+s]); __syncthreads(); }
    mx = red[0]; __syncthreads();
    float sum = 0.f;
    #pragma unroll
    for (int r = 0; r < 16; ++r) { v[r] = __expf(v[r] - mx); sum += v[r]; }
    red[t] = sum; __syncthreads();
    for (int s = 128; s > 0; s >>= 1) { if (t < s) red[t] += red[t+s]; __syncthreads(); }
    const float inv = 1.f/red[0];
    #pragma unroll
    for (int r = 0; r < 16; ++r) att[(size_t)b*HW_ + r*256 + t] = v[r]*inv;
}

// ---------------------------------------------------------------------------
// Kernel 4: out.flat[b,j] = att[b, j/128] * v_t.flat[b,j] + fmap.flat[b,j]
// (raw-reshape semantics of the reference), f16 v + float4 streaming
// ---------------------------------------------------------------------------
__global__ __launch_bounds__(256) void k_out(const float* __restrict__ fmap,
                                             const f16*   __restrict__ vt,
                                             const float* __restrict__ att,
                                             float* __restrict__ out) {
    const size_t i4 = ((size_t)blockIdx.x*256 + threadIdx.x)*4;
    const int b = (int)(i4 >> 19);             // / CHW_
    const int j = (int)(i4 & (CHW_ - 1));
    const float a = att[(size_t)b*HW_ + (j >> 7)];
    const v4h    vv = *(const v4h*)(vt + i4);
    const float4 fm = *(const float4*)(fmap + i4);
    float4 o;
    o.x = a*(float)vv[0] + fm.x; o.y = a*(float)vv[1] + fm.y;
    o.z = a*(float)vv[2] + fm.z; o.w = a*(float)vv[3] + fm.w;
    *(float4*)(out + i4) = o;
}

// ---------------------------------------------------------------------------
extern "C" void kernel_launch(void* const* d_in, const int* in_sizes, int n_in,
                              void* d_out, int out_size, void* d_ws, size_t ws_size,
                              hipStream_t stream) {
    const float* fmap = (const float*)d_in[0];   // (32,128,64,64) f32
    const float* W    = (const float*)d_in[1];   // (384,128) f32
    float* ws   = (float*)d_ws;
    float* cent = ws + WS_CENT;
    float* sim  = ws + WS_SIM;
    float* att  = ws + WS_ATT;
    f16*   Wh   = (f16*)(ws + WS_WH);
    f16*   vt   = (f16*)(ws + WS_VT);
    float* out  = (float*)d_out;

    k_wconv  <<<dim3(32),              dim3(256), 0, stream>>>(W, Wh);
    k_cent   <<<dim3(B_),              dim3(C_),  0, stream>>>(fmap, W, cent);
    k_qkv    <<<dim3(HW_/PTILE, B_),   dim3(256), 0, stream>>>(fmap, Wh, cent, sim, vt);
    k_softmax<<<dim3(B_),              dim3(256), 0, stream>>>(sim, att);
    k_out    <<<dim3((B_*CHW_)/1024),  dim3(256), 0, stream>>>(fmap, vt, att, out);
}
